// Model_9852654977721
// MI455X (gfx1250) — matrix operand order repaired and checked by reading, not compiled, not run on hardware
//
#include <hip/hip_runtime.h>

// ---------- types for WMMA ----------
typedef __attribute__((ext_vector_type(16))) __bf16 v16bf;
typedef __attribute__((ext_vector_type(8)))  __bf16 v8bf;
typedef __attribute__((ext_vector_type(8)))  float  v8f;

union ABfrag { v16bf v; v8bf h[2]; };

static __device__ __forceinline__ __bf16 f2bf(float f) {
    union { float f; unsigned u; } a; a.f = f;
    unsigned r = a.u + 0x7FFFu + ((a.u >> 16) & 1u);   // round-to-nearest-even
    unsigned short hs = (unsigned short)(r >> 16);
    __bf16 b;
    __builtin_memcpy(&b, &hs, sizeof(b));
    return b;
}

// =====================================================================
// Dual-input GEMM:  Out[M,64] = act( A0[M,:64]@W0 + A1[M,:64]@W1 + bias )
// K = 128 (two 64-wide halves).  128-row tiles, 8 waves, bf16 WMMA.
// =====================================================================
__global__ __launch_bounds__(256)
void k_dualgemm(const float* __restrict__ A0, int lda0,
                const float* __restrict__ A1, int lda1,
                const float* __restrict__ W0, const float* __restrict__ W1,
                const float* __restrict__ bias,
                float* __restrict__ Out, int M, int relu)
{
    constexpr int KP = 136;                 // 128 + 8 bf16 pad (row stride 272 B)
    __shared__ __bf16 sA[128 * KP];
    __shared__ __bf16 sB[64 * KP];          // W transposed: sB[n*KP + k]

    const int tid   = threadIdx.x;
    const int rowG0 = blockIdx.x * 128;

    // ---- stage A tile (fp32 -> bf16), float4 loads ----
    for (int idx = tid; idx < 128 * 32; idx += 256) {
        int row  = idx >> 5;
        int col  = (idx & 31) << 2;
        int grow = rowG0 + row;
        float4 v = make_float4(0.f, 0.f, 0.f, 0.f);
        if (grow < M) {
            const float* p = (col < 64) ? (A0 + (size_t)grow * lda0 + col)
                                        : (A1 + (size_t)grow * lda1 + (col - 64));
            v = *(const float4*)p;
        }
        __bf16* d = &sA[row * KP + col];
        d[0] = f2bf(v.x); d[1] = f2bf(v.y); d[2] = f2bf(v.z); d[3] = f2bf(v.w);
    }
    // ---- stage W (transposed, N-major) ----
    for (int idx = tid; idx < 128 * 64; idx += 256) {
        int k = idx >> 6;
        int n = idx & 63;
        float w = (k < 64) ? W0[k * 64 + n] : W1[(k - 64) * 64 + n];
        sB[n * KP + k] = f2bf(w);
    }
    __syncthreads();

    const int lane = tid & 31, wv = tid >> 5;
    const int g = lane >> 4, ln = lane & 15;
    const int r0 = wv * 16;                 // this wave's row strip
    v8f acc[4] = {};

#pragma unroll
    for (int kk = 0; kk < 128; kk += 32) {
        const v8bf* pa = (const v8bf*)&sA[(r0 + ln) * KP + kk + 8 * g];
        ABfrag a; a.h[0] = pa[0]; a.h[1] = pa[2];   // K = kk+8g..+7, kk+16+8g..+7
#pragma unroll
        for (int nt = 0; nt < 4; nt++) {
            const v8bf* pb = (const v8bf*)&sB[(nt * 16 + ln) * KP + kk + 8 * g];
            ABfrag b; b.h[0] = pb[0]; b.h[1] = pb[2];   // K = kk+8g..+7, kk+16+8g..+7
            acc[nt] = __builtin_amdgcn_wmma_f32_16x16x32_bf16(
                false, a.v, false, b.v, (short)0, acc[nt], false, false);
        }
    }

    const int rowBase = rowG0 + r0 + 8 * g;
#pragma unroll
    for (int nt = 0; nt < 4; nt++) {
        int col = nt * 16 + ln;
        float bv = bias[col];
#pragma unroll
        for (int r = 0; r < 8; r++) {
            int grow = rowBase + r;
            if (grow < M) {
                float v = acc[nt][r] + bv;
                if (relu) v = fmaxf(v, 0.f);
                Out[(size_t)grow * 64 + col] = v;
            }
        }
    }
}

// =====================================================================
// Degree:  deg[dst[e]] += 1
// =====================================================================
__global__ __launch_bounds__(256)
void k_deg(const long long* __restrict__ dst, float* __restrict__ deg, int E)
{
    int t = blockIdx.x * 256 + threadIdx.x;
    if (t < E) atomicAdd(&deg[(int)dst[t]], 1.0f);
}

// =====================================================================
// Scatter-add:  agg[dst[e],:] += feat[src[e],:] (+ efeat[e,:] if non-null)
// 16 threads/edge, 4 floats each.
// =====================================================================
__global__ __launch_bounds__(256)
void k_scatter(const long long* __restrict__ src, const long long* __restrict__ dst,
               const float* __restrict__ feat, const float* __restrict__ efeat,
               float* __restrict__ agg, int E)
{
    int t = blockIdx.x * 256 + threadIdx.x;
    if (t >= E * 16) return;
    int edge = t >> 4;
    int c    = (t & 15) << 2;
    int s = (int)src[edge];
    int d = (int)dst[edge];
    float4 v = *(const float4*)(feat + (size_t)s * 64 + c);
    if (efeat) {
        float4 ev = *(const float4*)(efeat + (size_t)edge * 64 + c);
        v.x += ev.x; v.y += ev.y; v.z += ev.z; v.w += ev.w;
    }
    float* a = agg + (size_t)d * 64 + c;
    atomicAdd(a + 0, v.x); atomicAdd(a + 1, v.y);
    atomicAdd(a + 2, v.z); atomicAdd(a + 3, v.w);
}

// =====================================================================
// agg[i,:] /= max(deg[i], 1)
// =====================================================================
__global__ __launch_bounds__(256)
void k_divdeg(float* __restrict__ agg, const float* __restrict__ deg, int Nn)
{
    int t = blockIdx.x * 256 + threadIdx.x;
    if (t < Nn * 64) agg[t] = agg[t] / fmaxf(deg[t >> 6], 1.0f);
}

// =====================================================================
// Fused edge predictor: gather concat([h2[src], h2[dst], e]) into LDS,
// K=192 WMMA GEMM -> relu -> z stays in LDS -> 64->1 dot -> logits.
// One block = 128 edges (E = 1.6M divides exactly).
// =====================================================================
__global__ __launch_bounds__(256)
void k_pred(const long long* __restrict__ src, const long long* __restrict__ dst,
            const float* __restrict__ h2, const float* __restrict__ efeat,
            const float* __restrict__ Wp1, const float* __restrict__ bp1,
            const float* __restrict__ Wp2, const float* __restrict__ bp2,
            float* __restrict__ out)
{
    constexpr int KP = 200;                 // 192 + 8 pad (row stride 400 B)
    __shared__ __bf16 sA[128 * KP];
    __shared__ __bf16 sB[64 * KP];
    __shared__ float  sZ[128 * 65];
    __shared__ int    sSrc[128], sDst[128];
    __shared__ float  sW2[64];

    const int tid = threadIdx.x;
    const int e0  = blockIdx.x * 128;

    if (tid < 128) { sSrc[tid] = (int)src[e0 + tid]; sDst[tid] = (int)dst[e0 + tid]; }
    if (tid >= 128 && tid < 192) sW2[tid - 128] = Wp2[tid - 128];
    // stage Wp1 transposed
    for (int idx = tid; idx < 192 * 64; idx += 256) {
        int k = idx >> 6, n = idx & 63;
        sB[n * KP + k] = f2bf(Wp1[k * 64 + n]);
    }
    __syncthreads();

    // gather-stage A tile: [h2[src] | h2[dst] | e]
    for (int idx = tid; idx < 128 * 48; idx += 256) {
        int row = idx / 48;
        int col = (idx % 48) << 2;
        const float* p;
        if (col < 64)       p = h2    + (size_t)sSrc[row] * 64 + col;
        else if (col < 128) p = h2    + (size_t)sDst[row] * 64 + (col - 64);
        else                p = efeat + (size_t)(e0 + row) * 64 + (col - 128);
        float4 v = *(const float4*)p;
        __bf16* d = &sA[row * KP + col];
        d[0] = f2bf(v.x); d[1] = f2bf(v.y); d[2] = f2bf(v.z); d[3] = f2bf(v.w);
    }
    __syncthreads();

    const int lane = tid & 31, wv = tid >> 5;
    const int g = lane >> 4, ln = lane & 15;
    const int r0 = wv * 16;
    v8f acc[4] = {};

#pragma unroll
    for (int kk = 0; kk < 192; kk += 32) {
        const v8bf* pa = (const v8bf*)&sA[(r0 + ln) * KP + kk + 8 * g];
        ABfrag a; a.h[0] = pa[0]; a.h[1] = pa[2];
#pragma unroll
        for (int nt = 0; nt < 4; nt++) {
            const v8bf* pb = (const v8bf*)&sB[(nt * 16 + ln) * KP + kk + 8 * g];
            ABfrag b; b.h[0] = pb[0]; b.h[1] = pb[2];
            acc[nt] = __builtin_amdgcn_wmma_f32_16x16x32_bf16(
                false, a.v, false, b.v, (short)0, acc[nt], false, false);
        }
    }

    const int rowBase = r0 + 8 * g;
#pragma unroll
    for (int nt = 0; nt < 4; nt++) {
        int col = nt * 16 + ln;
        float bv = bp1[col];
#pragma unroll
        for (int r = 0; r < 8; r++)
            sZ[(rowBase + r) * 65 + col] = fmaxf(acc[nt][r] + bv, 0.f);
    }
    __syncthreads();

    if (tid < 128) {
        float a = bp2[0];
        const float* zr = &sZ[tid * 65];
#pragma unroll 8
        for (int j = 0; j < 64; j++) a += zr[j] * sW2[j];
        out[e0 + tid] = a;
    }
}

// =====================================================================
// Host-side orchestration
// =====================================================================
extern "C" void kernel_launch(void* const* d_in, const int* in_sizes, int n_in,
                              void* d_out, int out_size, void* d_ws, size_t ws_size,
                              hipStream_t stream)
{
    (void)in_sizes; (void)n_in; (void)out_size; (void)ws_size;

    const float*     NF  = (const float*)d_in[0];
    const float*     EF  = (const float*)d_in[1];
    const long long* EI  = (const long long*)d_in[2];
    const float*     Wnr = (const float*)d_in[3];
    const float*     bnr = (const float*)d_in[4];
    const float*     Wer = (const float*)d_in[5];
    const float*     ber = (const float*)d_in[6];
    const float*     Ws1 = (const float*)d_in[7];
    const float*     Wn1 = (const float*)d_in[8];
    const float*     b1  = (const float*)d_in[9];
    const float*     Ws2 = (const float*)d_in[10];
    const float*     Wn2 = (const float*)d_in[11];
    const float*     b2  = (const float*)d_in[12];
    const float*     Wp1 = (const float*)d_in[13];
    const float*     bp1 = (const float*)d_in[14];
    const float*     Wp2 = (const float*)d_in[15];
    const float*     bp2 = (const float*)d_in[16];
    float* out = (float*)d_out;

    const long long Nn = 100000, Ee = 1600000;
    const long long* srcp = EI;
    const long long* dstp = EI + Ee;

    // workspace layout (floats):
    float* ws     = (float*)d_ws;
    float* e_feat = ws;                     // Ee*64  (409.6 MB)
    float* n_feat = e_feat + Ee * 64;       // Nn*64  (reused as agg2)
    float* agg1   = n_feat + Nn * 64;       // Nn*64  (reused as h2)
    float* h      = agg1   + Nn * 64;       // Nn*64
    float* deg    = h      + Nn * 64;       // Nn
    float* agg2   = n_feat;
    float* h2     = agg1;

    hipMemsetAsync(deg,  0, Nn * sizeof(float),      stream);
    hipMemsetAsync(agg1, 0, Nn * 64 * sizeof(float), stream);

    dim3 blk(256);
    const int gbN = (int)((Nn + 127) / 128);
    const int gbE = (int)((Ee + 127) / 128);

    // feature reducers (K=128 split into two 64 halves of the same matrix)
    k_dualgemm<<<gbN, blk, 0, stream>>>(NF, 128, NF + 64, 128,
                                        Wnr, Wnr + 64 * 64, bnr, n_feat, (int)Nn, 1);
    k_dualgemm<<<gbE, blk, 0, stream>>>(EF, 128, EF + 64, 128,
                                        Wer, Wer + 64 * 64, ber, e_feat, (int)Ee, 1);
    // degree + layer-1 aggregation
    k_deg<<<(int)((Ee + 255) / 256), blk, 0, stream>>>(dstp, deg, (int)Ee);
    k_scatter<<<(int)((Ee * 16 + 255) / 256), blk, 0, stream>>>(srcp, dstp, n_feat,
                                                                e_feat, agg1, (int)Ee);
    k_divdeg<<<(int)((Nn * 64 + 255) / 256), blk, 0, stream>>>(agg1, deg, (int)Nn);
    // SAGE layer 1 (fused dual GEMM, relu)
    k_dualgemm<<<gbN, blk, 0, stream>>>(n_feat, 64, agg1, 64,
                                        Ws1, Wn1, b1, h, (int)Nn, 1);
    // layer-2 aggregation (agg2 reuses n_feat slot)
    hipMemsetAsync(agg2, 0, Nn * 64 * sizeof(float), stream);
    k_scatter<<<(int)((Ee * 16 + 255) / 256), blk, 0, stream>>>(srcp, dstp, h,
                                                                nullptr, agg2, (int)Ee);
    k_divdeg<<<(int)((Nn * 64 + 255) / 256), blk, 0, stream>>>(agg2, deg, (int)Nn);
    // SAGE layer 2 (no relu); h2 reuses agg1 slot
    k_dualgemm<<<gbN, blk, 0, stream>>>(h, 64, agg2, 64,
                                        Ws2, Wn2, b2, h2, (int)Nn, 0);
    // fused predictor
    k_pred<<<(int)(Ee / 128), blk, 0, stream>>>(srcp, dstp, h2, e_feat,
                                                Wp1, bp1, Wp2, bp2, out);
}
